// NeuralMemory_5471788335888
// MI455X (gfx1250) — matrix operand order repaired and checked by reading, not compiled, not run on hardware
//
#include <hip/hip_runtime.h>
#include <hip/hip_bf16.h>

typedef __attribute__((ext_vector_type(16))) _Float16 v16h;
typedef __attribute__((ext_vector_type(8)))  _Float16 v8h;
typedef __attribute__((ext_vector_type(8)))  float    v8f;
typedef __attribute__((ext_vector_type(4)))  float    v4f;
typedef __attribute__((ext_vector_type(4)))  int      v4i;

// Probe for the gfx1250 async global->LDS copy builtin; fall back to the
// synchronous VGPR-mediated path if this toolchain doesn't declare it.
// Probe-learned signature: (int4 __device__* gsrc, int4 __shared__* ldst, int, int)
#if defined(__has_builtin)
#if __has_builtin(__builtin_amdgcn_global_load_async_to_lds_b128)
#define HAVE_ASYNC_LDS 1
#endif
#endif
#ifndef HAVE_ASYNC_LDS
#define HAVE_ASYNC_LDS 0
#endif

typedef __attribute__((address_space(1))) v4i* gp_v4i;
typedef __attribute__((address_space(3))) v4i* lp_v4i;

__device__ __forceinline__ void wait_async_zero() {
#if defined(__has_builtin)
#if __has_builtin(__builtin_amdgcn_s_wait_asynccnt)
  __builtin_amdgcn_s_wait_asynccnt(0);
#else
  asm volatile("s_wait_asynccnt 0" ::: "memory");
#endif
#else
  asm volatile("s_wait_asynccnt 0" ::: "memory");
#endif
}

// Problem constants (match the reference)
#define KD   512      // D == H
#define KT   2112     // M + S tokens per batch
#define KB   2
#define KS   2048
#define KM   64
#define KNC  16
#define KNH  8
#define KWIN 256

__device__ __forceinline__ float siluf(float x) { return x / (1.f + __expf(-x)); }

__device__ __forceinline__ v16h mkfrag16(const _Float16* p0, const _Float16* p1) {
  v8h lo = *(const v8h*)p0;
  v8h hi = *(const v8h*)p1;
  v16h f;
#pragma unroll
  for (int j = 0; j < 8; ++j) { f[j] = lo[j]; f[j + 8] = hi[j]; }
  return f;
}

// ------------------------------------------------------------------
// Generic f16-WMMA GEMM: C[Nr x Nc] = op(A) @ op(B) (+bias), fp32 accum.
// OPA==0: A is [Nr x Kd] row-major. OPA==1: A is [Kd x Nr] row-major (A^T used).
// OPB==0: B is [Kd x Nc] row-major. OPB==1: B is [Nc x Kd] row-major (B^T used).
// EPI: 0 = acc+bias ; 1 = silu(acc+bias) ; 2 = res + silu(acc+bias) ; 3 = res + acc
// Requires Nr,Nc % 64 == 0, Kd % 32 == 0 (true for every call here).
// OPA==0 stages the A tile with GLOBAL_LOAD_ASYNC_TO_LDS_B128 (fp32 in LDS,
// converted to f16 during fragment build); the transposing stages (B, OPA==1)
// use the VGPR path.
// ------------------------------------------------------------------
template <int OPA, int OPB, int EPI>
__global__ __launch_bounds__(256) void gemm_wmma_kernel(
    const float* __restrict__ A, const float* __restrict__ Bm,
    const float* __restrict__ bias, const float* __restrict__ Rs,
    float* __restrict__ C, int Nr, int Nc, int Kd) {
  constexpr int LDT = 40;  // halfs per LDS row (80B): keeps 16B-aligned segments
#if HAVE_ASYNC_LDS
  constexpr int LDAF = 36; // floats per LDS row (144B) for the async fp32 A tile
  __shared__ __align__(16) float sAF[64 * LDAF];
#endif
  __shared__ __align__(16) _Float16 sA[64 * LDT];  // [m][k]
  __shared__ __align__(16) _Float16 sB[64 * LDT];  // [n][k]  (B^T tile)
  const int tid  = threadIdx.x;
  const int lane = tid & 31;
  const int wave = tid >> 5;
  const int wm   = wave & 3;    // 16-row band
  const int wn   = wave >> 2;   // 32-col band (2 accumulators)
  const int row0 = blockIdx.y * 64;
  const int col0 = blockIdx.x * 64;
  const int r    = lane & 15;
  const int hh   = lane >> 4;
  v8f acc[2] = {};

  for (int k0 = 0; k0 < Kd; k0 += 32) {
    __syncthreads();
    if (OPA == 0) {
#if HAVE_ASYNC_LDS
      // 64x32 fp32 tile = 512 16B chunks; 256 threads x 2 async copies.
      const int c0 = tid << 1;
#pragma unroll
      for (int u = 0; u < 2; ++u) {
        const int c = c0 + u;
        const int ar = c >> 3, af4 = (c & 7) << 2;
        const float* g = A + (size_t)(row0 + ar) * Kd + (k0 + af4);
        __builtin_amdgcn_global_load_async_to_lds_b128(
            (gp_v4i)g, (lp_v4i)(sAF + ar * LDAF + af4), 0, 0);
      }
#else
      const int ar = tid >> 2, ak = (tid & 3) << 3;
      const float* g = A + (size_t)(row0 + ar) * Kd + (k0 + ak);
      v4f x0 = *(const v4f*)g, x1 = *(const v4f*)(g + 4);
      _Float16* d = sA + ar * LDT + ak;
#pragma unroll
      for (int j = 0; j < 4; ++j) { d[j] = (_Float16)x0[j]; d[4 + j] = (_Float16)x1[j]; }
#endif
    } else {
      const int ak = tid >> 3, ai = (tid & 7) << 3;
      const float* g = A + (size_t)(k0 + ak) * Nr + (row0 + ai);
      v4f x0 = *(const v4f*)g, x1 = *(const v4f*)(g + 4);
#pragma unroll
      for (int j = 0; j < 4; ++j) {
        sA[(ai + j) * LDT + ak]     = (_Float16)x0[j];
        sA[(ai + 4 + j) * LDT + ak] = (_Float16)x1[j];
      }
    }
    if (OPB == 0) {
      const int bk = tid >> 3, bj = (tid & 7) << 3;
      const float* g = Bm + (size_t)(k0 + bk) * Nc + (col0 + bj);
      v4f x0 = *(const v4f*)g, x1 = *(const v4f*)(g + 4);
#pragma unroll
      for (int j = 0; j < 4; ++j) {
        sB[(bj + j) * LDT + bk]     = (_Float16)x0[j];
        sB[(bj + 4 + j) * LDT + bk] = (_Float16)x1[j];
      }
    } else {
      const int bj = tid >> 2, bk = (tid & 3) << 3;
      const float* g = Bm + (size_t)(col0 + bj) * Kd + (k0 + bk);
      v4f x0 = *(const v4f*)g, x1 = *(const v4f*)(g + 4);
      _Float16* d = sB + bj * LDT + bk;
#pragma unroll
      for (int j = 0; j < 4; ++j) { d[j] = (_Float16)x0[j]; d[4 + j] = (_Float16)x1[j]; }
    }
#if HAVE_ASYNC_LDS
    if (OPA == 0) wait_async_zero();  // drain this wave's async copies
#endif
    __syncthreads();

    // A fragment: lane r holds row r; K halves {hh*8..+7, 16+hh*8..+7}
    v16h af;
    if (OPA == 0) {
#if HAVE_ASYNC_LDS
      const float* paf = sAF + (wm * 16 + r) * LDAF;
      v4f f0 = *(const v4f*)(paf + hh * 8);
      v4f f1 = *(const v4f*)(paf + hh * 8 + 4);
      v4f f2 = *(const v4f*)(paf + 16 + hh * 8);
      v4f f3 = *(const v4f*)(paf + 16 + hh * 8 + 4);
#pragma unroll
      for (int j = 0; j < 4; ++j) {
        af[j]      = (_Float16)f0[j];
        af[j + 4]  = (_Float16)f1[j];
        af[j + 8]  = (_Float16)f2[j];
        af[j + 12] = (_Float16)f3[j];
      }
#else
      const _Float16* pa = sA + (wm * 16 + r) * LDT;
      af = mkfrag16(pa + hh * 8, pa + 16 + hh * 8);
#endif
    } else {
      const _Float16* pa = sA + (wm * 16 + r) * LDT;
      af = mkfrag16(pa + hh * 8, pa + 16 + hh * 8);
    }
#pragma unroll
    for (int t = 0; t < 2; ++t) {
      // B fragment: lane r holds col r; K halves {hh*8..+7, 16+hh*8..+7}
      const _Float16* pb = sB + (wn * 32 + t * 16 + r) * LDT;
      v16h bf = mkfrag16(pb + hh * 8, pb + 16 + hh * 8);
      acc[t] = __builtin_amdgcn_wmma_f32_16x16x32_f16(false, af, false, bf,
                                                      (short)0, acc[t], false, false);
    }
  }

#pragma unroll
  for (int t = 0; t < 2; ++t) {
    const int col = col0 + wn * 32 + t * 16 + r;
    const float bcol = (bias != nullptr) ? bias[col] : 0.f;
#pragma unroll
    for (int v = 0; v < 8; ++v) {
      const int row = row0 + wm * 16 + v + 8 * hh;
      const size_t idx = (size_t)row * Nc + col;
      float y = acc[t][v] + bcol;
      if (EPI == 1) y = siluf(y);
      else if (EPI == 2) y = Rs[idx] + siluf(y);
      else if (EPI == 3) y = Rs[idx] + y;
      C[idx] = y;
    }
  }
}

// ------------------------------------------------------------------
// Sliding-window attention (flash-style, WMMA for QK^T and PV).
// Q/K/V/O are [B*T, 512] with head h in cols [h*64, h*64+64).
// Block: 128 threads (4 waves), grid (T/64, NH, B). Wave w owns 16 queries.
// ------------------------------------------------------------------
__global__ __launch_bounds__(128) void swa_attn_kernel(
    const float* __restrict__ Q, const float* __restrict__ Km,
    const float* __restrict__ V, float* __restrict__ O, int T) {
  constexpr int LQ = 72;  // half stride (144B rows, 16B-aligned segments)
  __shared__ __align__(16) _Float16 sQ[64 * LQ];   // [q][d]
  __shared__ __align__(16) _Float16 sK[64 * LQ];   // [key][d]
  __shared__ __align__(16) _Float16 sV[64 * LQ];   // [d][key] (transposed)
  __shared__ float sS[4][16][68];                  // per-wave score tile
  __shared__ __align__(16) _Float16 sP[4][16 * LQ];// per-wave P tile
  __shared__ float sM[4][16], sL[4][16], sF[4][16];

  const int tid  = threadIdx.x;
  const int lane = tid & 31;
  const int wave = tid >> 5;
  const int q0   = blockIdx.x * 64;
  const int head = blockIdx.y;
  const int b    = blockIdx.z;
  const size_t base = ((size_t)b * T) * KD + head * 64;
  const int r  = lane & 15;
  const int hh = lane >> 4;
  const int iq = q0 + wave * 16;
  const float scale = 0.125f;  // 1/sqrt(64)

  {  // stage Q tile (fp32 -> f16)
    const int row = tid >> 1, c0 = (tid & 1) * 32;
    const float* g = Q + base + (size_t)(q0 + row) * KD + c0;
#pragma unroll
    for (int u = 0; u < 32; u += 4) {
      v4f x = *(const v4f*)(g + u);
#pragma unroll
      for (int w2 = 0; w2 < 4; ++w2) sQ[row * LQ + c0 + u + w2] = (_Float16)x[w2];
    }
  }
  if (lane < 16) { sM[wave][lane] = -1e30f; sL[wave][lane] = 0.f; }

  v8f acc[4] = {};

  for (int kt = 0; kt < 5; ++kt) {
    const int jb = q0 - 256 + kt * 64;       // tile base is a multiple of 64
    if (jb + 63 < 0) continue;               // fully out of range (uniform)
    __syncthreads();
    {  // stage K and V tiles
      const int key = tid >> 1, c0 = (tid & 1) * 32;
      const int j = jb + key;
      if (j >= 0 && j < T) {
        const float* gk = Km + base + (size_t)j * KD + c0;
        const float* gv = V  + base + (size_t)j * KD + c0;
#pragma unroll
        for (int u = 0; u < 32; u += 4) {
          v4f xk = *(const v4f*)(gk + u);
          v4f xv = *(const v4f*)(gv + u);
#pragma unroll
          for (int w2 = 0; w2 < 4; ++w2) {
            sK[key * LQ + c0 + u + w2]   = (_Float16)xk[w2];
            sV[(c0 + u + w2) * LQ + key] = (_Float16)xv[w2];
          }
        }
      } else {
#pragma unroll
        for (int u = 0; u < 32; ++u) {
          sK[key * LQ + c0 + u]   = (_Float16)0.f;
          sV[(c0 + u) * LQ + key] = (_Float16)0.f;
        }
      }
    }
    __syncthreads();

    // scores S = q @ k^T  (16 x 64, K-dim = 64 -> two 32-wide steps)
    v8f sc[4] = {};
#pragma unroll
    for (int ks = 0; ks < 2; ++ks) {
      const _Float16* pa = sQ + (wave * 16 + r) * LQ + ks * 32;
      v16h af = mkfrag16(pa + hh * 8, pa + 16 + hh * 8);
#pragma unroll
      for (int t = 0; t < 4; ++t) {
        const _Float16* pb = sK + (t * 16 + r) * LQ + ks * 32;
        v16h bf = mkfrag16(pb + hh * 8, pb + 16 + hh * 8);
        sc[t] = __builtin_amdgcn_wmma_f32_16x16x32_f16(false, af, false, bf,
                                                       (short)0, sc[t], false, false);
      }
    }
#pragma unroll
    for (int t = 0; t < 4; ++t)
#pragma unroll
      for (int v = 0; v < 8; ++v) sS[wave][v + 8 * hh][t * 16 + r] = sc[t][v];
    __syncthreads();

    {  // masked online softmax; lane pair (seg 0/1) owns one query row
      const int rr = lane & 15, seg = lane >> 4;
      const int i = iq + rr;
      float mloc = -1e30f;
#pragma unroll
      for (int c = 0; c < 32; ++c) {
        const int cc = seg * 32 + c;
        const int j = jb + cc;
        const bool valid = (j >= 0) && (j <= i) && (i - j < KWIN);
        const float s = sS[wave][rr][cc] * scale;
        if (valid) mloc = fmaxf(mloc, s);
      }
      const float mt   = fmaxf(mloc, __shfl_xor(mloc, 16, 32));
      const float mold = sM[wave][rr];
      const float mnew = fmaxf(mold, mt);
      const float fac  = __expf(mold - mnew);
      float lloc = 0.f;
#pragma unroll
      for (int c = 0; c < 32; ++c) {
        const int cc = seg * 32 + c;
        const int j = jb + cc;
        const bool valid = (j >= 0) && (j <= i) && (i - j < KWIN);
        const float p = valid ? __expf(sS[wave][rr][cc] * scale - mnew) : 0.f;
        sP[wave][rr * LQ + cc] = (_Float16)p;
        lloc += p;
      }
      lloc += __shfl_xor(lloc, 16, 32);
      if (seg == 0) {
        sM[wave][rr] = mnew;
        sL[wave][rr] = sL[wave][rr] * fac + lloc;
        sF[wave][rr] = fac;
      }
    }
    __syncthreads();

    // rescale running output, then out += P @ V
#pragma unroll
    for (int t = 0; t < 4; ++t)
#pragma unroll
      for (int v = 0; v < 8; ++v) acc[t][v] *= sF[wave][v + 8 * hh];
#pragma unroll
    for (int ks = 0; ks < 2; ++ks) {
      const _Float16* pa = sP[wave] + r * LQ + ks * 32;
      v16h af = mkfrag16(pa + hh * 8, pa + 16 + hh * 8);
#pragma unroll
      for (int t = 0; t < 4; ++t) {
        const _Float16* pb = sV + (t * 16 + r) * LQ + ks * 32;
        v16h bf = mkfrag16(pb + hh * 8, pb + 16 + hh * 8);
        acc[t] = __builtin_amdgcn_wmma_f32_16x16x32_f16(false, af, false, bf,
                                                        (short)0, acc[t], false, false);
      }
    }
  }
  __syncthreads();
#pragma unroll
  for (int t = 0; t < 4; ++t)
#pragma unroll
    for (int v = 0; v < 8; ++v) {
      const int row = iq + v + 8 * hh;
      const float l = sL[wave][v + 8 * hh];
      O[base + (size_t)row * KD + t * 16 + r] = acc[t][v] / l;
    }
}

// ----------------------- small support kernels -----------------------

__global__ void build_xm_kernel(const float* __restrict__ x,
                                const float* __restrict__ meta,
                                float* __restrict__ xm) {
  const size_t n = (size_t)KB * KT * KD;
  size_t i = (size_t)blockIdx.x * 256 + threadIdx.x;
  if (i >= n) return;
  const int d = (int)(i % KD);
  const size_t td = i / KD;
  const int t = (int)(td % KT);
  const int b = (int)(td / KT);
  xm[i] = (t < KM) ? meta[(size_t)t * KD + d]
                   : x[((size_t)b * KS + (t - KM)) * KD + d];
}

// y[b,t,d] = silu( sum_tau w[d,tau] * xin[b,t-3+tau,d] ), causal
__global__ void conv_silu_kernel(const float* __restrict__ Xin,
                                 const float* __restrict__ w,
                                 float* __restrict__ Y) {
  const size_t n = (size_t)KB * KT * KD;
  size_t i = (size_t)blockIdx.x * 256 + threadIdx.x;
  if (i >= n) return;
  const int d = (int)(i % KD);
  const size_t td = i / KD;
  const int t = (int)(td % KT);
  const size_t bt0 = td - t;  // b*T
  float a = 0.f;
#pragma unroll
  for (int tau = 0; tau < 4; ++tau) {
    const int tt = t - 3 + tau;
    if (tt >= 0) a += Xin[(bt0 + tt) * KD + d] * w[d * 4 + tau];
  }
  Y[i] = siluf(a);
}

// per-token l2 normalize over D=512 (in place); one block / token
__global__ __launch_bounds__(128) void l2norm_kernel(float* __restrict__ X) {
  __shared__ float red[128];
  const size_t base = (size_t)blockIdx.x * KD;
  float s = 0.f;
  for (int i = threadIdx.x; i < KD; i += 128) { const float v = X[base + i]; s += v * v; }
  red[threadIdx.x] = s;
  __syncthreads();
  for (int off = 64; off > 0; off >>= 1) {
    if (threadIdx.x < off) red[threadIdx.x] += red[threadIdx.x + off];
    __syncthreads();
  }
  const float inv = 1.f / fmaxf(sqrtf(red[0]), 1e-12f);
  for (int i = threadIdx.x; i < KD; i += 128) X[base + i] *= inv;
}

// gate[t] = maxw * sigmoid( x[t,:] . w + b ); one wave per token
__global__ __launch_bounds__(256) void gate_kernel(const float* __restrict__ X,
                                                   const float* __restrict__ w,
                                                   const float* __restrict__ b,
                                                   float* __restrict__ out,
                                                   float maxw, int ntok) {
  const int wave = threadIdx.x >> 5, lane = threadIdx.x & 31;
  const int t = blockIdx.x * 8 + wave;
  if (t >= ntok) return;
  const float* x = X + (size_t)t * KD;
  float s = 0.f;
  for (int i = lane; i < KD; i += 32) s += x[i] * w[i];
#pragma unroll
  for (int off = 16; off > 0; off >>= 1) s += __shfl_xor(s, off, 32);
  if (lane == 0) out[t] = maxw / (1.f + __expf(-(s + b[0])));
}

// scal = {theta=mean(lr), alpha=mean(fg), eta=mean(mo)}
__global__ __launch_bounds__(256) void means_kernel(const float* __restrict__ lr,
                                                    const float* __restrict__ fg,
                                                    const float* __restrict__ mo,
                                                    float* __restrict__ scal, int n) {
  __shared__ float r0[256], r1[256], r2[256];
  float s0 = 0.f, s1 = 0.f, s2 = 0.f;
  for (int i = threadIdx.x; i < n; i += 256) { s0 += lr[i]; s1 += fg[i]; s2 += mo[i]; }
  r0[threadIdx.x] = s0; r1[threadIdx.x] = s1; r2[threadIdx.x] = s2;
  __syncthreads();
  for (int off = 128; off > 0; off >>= 1) {
    if (threadIdx.x < off) {
      r0[threadIdx.x] += r0[threadIdx.x + off];
      r1[threadIdx.x] += r1[threadIdx.x + off];
      r2[threadIdx.x] += r2[threadIdx.x + off];
    }
    __syncthreads();
  }
  if (threadIdx.x == 0) { scal[0] = r0[0] / n; scal[1] = r1[0] / n; scal[2] = r2[0] / n; }
}

__global__ void silu_kernel(const float* __restrict__ S, float* __restrict__ H, size_t n) {
  size_t i = (size_t)blockIdx.x * 256 + threadIdx.x;
  if (i < n) H[i] = siluf(S[i]);
}
__global__ void res_silu_kernel(const float* __restrict__ R, const float* __restrict__ S,
                                float* __restrict__ H, size_t n) {
  size_t i = (size_t)blockIdx.x * 256 + threadIdx.x;
  if (i < n) H[i] = R[i] + siluf(S[i]);
}
// out = dh * silu'(s)
__global__ void dsilu_kernel(const float* __restrict__ DH, const float* __restrict__ S,
                             float* __restrict__ OUT, size_t n) {
  size_t i = (size_t)blockIdx.x * 256 + threadIdx.x;
  if (i < n) {
    const float s = S[i];
    const float sg = 1.f / (1.f + __expf(-s));
    OUT[i] = DH[i] * sg * (1.f + s * (1.f - sg));
  }
}
// dpred = (2/(D*NC)) * lr_row * (pred - v)
__global__ void dpred_kernel(const float* __restrict__ P, const float* __restrict__ Vv,
                             const float* __restrict__ lr, float* __restrict__ OUT, size_t n) {
  size_t i = (size_t)blockIdx.x * 256 + threadIdx.x;
  if (i < n) OUT[i] = (2.f / (512.f * 16.f)) * lr[i >> 9] * (P[i] - Vv[i]);
}
// g_b[j] = sum_rows X[:, j]
__global__ void colsum_kernel(const float* __restrict__ X, float* __restrict__ out,
                              int Nr, int Nc) {
  const int c = blockIdx.x * blockDim.x + threadIdx.x;
  if (c >= Nc) return;
  float s = 0.f;
  for (int rr = 0; rr < Nr; ++rr) s += X[(size_t)rr * Nc + c];
  out[c] = s;
}
// new_p = p*(1-alpha) + eta*mom - theta*g
__global__ void update_kernel(const float* __restrict__ p, const float* __restrict__ mom,
                              const float* __restrict__ g, const float* __restrict__ scal,
                              float* __restrict__ out, size_t n) {
  size_t i = (size_t)blockIdx.x * 256 + threadIdx.x;
  if (i < n) out[i] = p[i] * (1.f - scal[1]) + scal[2] * mom[i] - scal[0] * g[i];
}
__global__ void slice_out_kernel(const float* __restrict__ Xin, float* __restrict__ out) {
  const size_t n = (size_t)KB * KS * KD;
  size_t i = (size_t)blockIdx.x * 256 + threadIdx.x;
  if (i >= n) return;
  const int d = (int)(i % KD);
  const size_t sd = i / KD;
  const int s = (int)(sd % KS);
  const int b = (int)(sd / KS);
  out[i] = Xin[((size_t)b * KT + KM + s) * KD + d];
}

// ------------------------------------------------------------------

extern "C" void kernel_launch(void* const* d_in, const int* in_sizes, int n_in,
                              void* d_out, int out_size, void* d_ws, size_t ws_size,
                              hipStream_t stream) {
  (void)in_sizes; (void)n_in; (void)out_size; (void)ws_size;
  const float* x         = (const float*)d_in[0];
  const float* meta      = (const float*)d_in[1];
  const float* mem_w_in  = (const float*)d_in[2];
  const float* mem_b_in  = (const float*)d_in[3];
  const float* mem_w_h   = (const float*)d_in[4];
  const float* mem_b_h   = (const float*)d_in[5];
  const float* mem_w_out = (const float*)d_in[6];
  const float* mem_b_out = (const float*)d_in[7];
  const float* mom_w_in  = (const float*)d_in[8];
  const float* mom_b_in  = (const float*)d_in[9];
  const float* mom_w_h   = (const float*)d_in[10];
  const float* mom_b_h   = (const float*)d_in[11];
  const float* mom_w_out = (const float*)d_in[12];
  const float* mom_b_out = (const float*)d_in[13];
  const float* q_w  = (const float*)d_in[14];
  const float* q_b  = (const float*)d_in[15];
  const float* q_cv = (const float*)d_in[16];
  const float* k_w  = (const float*)d_in[17];
  const float* k_b  = (const float*)d_in[18];
  const float* k_cv = (const float*)d_in[19];
  const float* v_w  = (const float*)d_in[20];
  const float* v_b  = (const float*)d_in[21];
  const float* v_cv = (const float*)d_in[22];
  const float* lr_w = (const float*)d_in[23];
  const float* lr_b = (const float*)d_in[24];
  const float* fg_w = (const float*)d_in[25];
  const float* fg_b = (const float*)d_in[26];
  const float* mo_w = (const float*)d_in[27];
  const float* mo_b = (const float*)d_in[28];
  const float* swa_wq = (const float*)d_in[29];
  const float* swa_wk = (const float*)d_in[30];
  const float* swa_wv = (const float*)d_in[31];
  const float* swa_wo = (const float*)d_in[32];
  const float* swa_bq = (const float*)d_in[33];
  const float* swa_bk = (const float*)d_in[34];
  const float* swa_bv = (const float*)d_in[35];
  const float* swa_bo = (const float*)d_in[36];

  const int    N  = KB * KT;           // 4224 tokens
  const size_t ND = (size_t)N * KD;    // token activations
  const size_t PW = (size_t)KD * KD;   // 512x512 weight

  float* w = (float*)d_ws;
  size_t off = 0;
  auto alloc = [&](size_t nfl) { float* p = w + off; off += (nfl + 63) & ~(size_t)63; return p; };
  float* xm  = alloc(ND);
  float* qn  = alloc(ND);
  float* kn  = alloc(ND);
  float* vv  = alloc(ND);
  float* lrg = alloc(N);
  float* fgg = alloc(N);
  float* mog = alloc(N);
  float* s_in = alloc(ND);
  float* h0   = alloc(ND);
  float* s0   = alloc(ND);
  float* h1   = alloc(ND);
  float* s1   = alloc(ND);
  float* h2   = alloc(ND);
  float* pr   = alloc(ND);
  float* d1   = alloc(ND);
  float* d2   = alloc(ND);
  float* gwin  = alloc(PW); float* gbin  = alloc(KD);
  float* gwh0  = alloc(PW); float* gbh0  = alloc(KD);
  float* gwh1  = alloc(PW); float* gbh1  = alloc(KD);
  float* gwout = alloc(PW); float* gbout = alloc(KD);
  float* nwin  = alloc(PW); float* nbin  = alloc(KD);
  float* nwh0  = alloc(PW); float* nbh0  = alloc(KD);
  float* nwh1  = alloc(PW); float* nbh1  = alloc(KD);
  float* nwout = alloc(PW); float* nbout = alloc(KD);
  float* scal  = alloc(64);

  auto gg = [](int Nr, int Nc) { return dim3((unsigned)(Nc / 64), (unsigned)(Nr / 64), 1); };
  auto eb = [](size_t nfl) { return dim3((unsigned)((nfl + 255) / 256), 1, 1); };
  const dim3 b256(256), b128(128);

  // ---- stage 0: concat meta memory + x
  build_xm_kernel<<<eb(ND), b256, 0, stream>>>(x, meta, xm);

  // ---- stage 1: q/k/v = l2n?(silu(causal_dwconv(xm @ W + b)))
  gemm_wmma_kernel<0,0,0><<<gg(N,KD), b256, 0, stream>>>(xm, q_w, q_b, nullptr, d1, N, KD, KD);
  conv_silu_kernel<<<eb(ND), b256, 0, stream>>>(d1, q_cv, qn);
  l2norm_kernel<<<dim3((unsigned)N), b128, 0, stream>>>(qn);
  gemm_wmma_kernel<0,0,0><<<gg(N,KD), b256, 0, stream>>>(xm, k_w, k_b, nullptr, d1, N, KD, KD);
  conv_silu_kernel<<<eb(ND), b256, 0, stream>>>(d1, k_cv, kn);
  l2norm_kernel<<<dim3((unsigned)N), b128, 0, stream>>>(kn);
  gemm_wmma_kernel<0,0,0><<<gg(N,KD), b256, 0, stream>>>(xm, v_w, v_b, nullptr, d1, N, KD, KD);
  conv_silu_kernel<<<eb(ND), b256, 0, stream>>>(d1, v_cv, vv);

  // ---- stage 2: gates + global means
  gate_kernel<<<dim3((unsigned)(N / 8)), b256, 0, stream>>>(xm, lr_w, lr_b, lrg, 0.1f, N);
  gate_kernel<<<dim3((unsigned)(N / 8)), b256, 0, stream>>>(xm, fg_w, fg_b, fgg, 1.0f, N);
  gate_kernel<<<dim3((unsigned)(N / 8)), b256, 0, stream>>>(xm, mo_w, mo_b, mog, 1.0f, N);
  means_kernel<<<dim3(1), b256, 0, stream>>>(lrg, fgg, mog, scal, N);

  // ---- stage 3: MLP forward (keep pre-activations for backward)
  const float* w_h0 = mem_w_h;            const float* w_h1 = mem_w_h + PW;
  const float* b_h0 = mem_b_h;            const float* b_h1 = mem_b_h + KD;
  gemm_wmma_kernel<0,0,0><<<gg(N,KD), b256, 0, stream>>>(kn, mem_w_in, mem_b_in, nullptr, s_in, N, KD, KD);
  silu_kernel<<<eb(ND), b256, 0, stream>>>(s_in, h0, ND);
  gemm_wmma_kernel<0,0,0><<<gg(N,KD), b256, 0, stream>>>(h0, w_h0, b_h0, nullptr, s0, N, KD, KD);
  res_silu_kernel<<<eb(ND), b256, 0, stream>>>(h0, s0, h1, ND);
  gemm_wmma_kernel<0,0,0><<<gg(N,KD), b256, 0, stream>>>(h1, w_h1, b_h1, nullptr, s1, N, KD, KD);
  res_silu_kernel<<<eb(ND), b256, 0, stream>>>(h1, s1, h2, ND);
  gemm_wmma_kernel<0,0,0><<<gg(N,KD), b256, 0, stream>>>(h2, mem_w_out, mem_b_out, nullptr, pr, N, KD, KD);

  // ---- stage 4: backward -> weight/bias gradients
  dpred_kernel<<<eb(ND), b256, 0, stream>>>(pr, vv, lrg, d1, ND);               // dpred
  gemm_wmma_kernel<1,0,0><<<gg(KD,KD), b256, 0, stream>>>(h2, d1, nullptr, nullptr, gwout, KD, KD, N);
  colsum_kernel<<<dim3(2), b256, 0, stream>>>(d1, gbout, N, KD);
  gemm_wmma_kernel<0,1,0><<<gg(N,KD), b256, 0, stream>>>(d1, mem_w_out, nullptr, nullptr, d2, N, KD, KD); // dh2
  dsilu_kernel<<<eb(ND), b256, 0, stream>>>(d2, s1, d1, ND);                     // ds1
  gemm_wmma_kernel<1,0,0><<<gg(KD,KD), b256, 0, stream>>>(h1, d1, nullptr, nullptr, gwh1, KD, KD, N);
  colsum_kernel<<<dim3(2), b256, 0, stream>>>(d1, gbh1, N, KD);
  gemm_wmma_kernel<0,1,3><<<gg(N,KD), b256, 0, stream>>>(d1, w_h1, nullptr, d2, pr, N, KD, KD); // dh1
  dsilu_kernel<<<eb(ND), b256, 0, stream>>>(pr, s0, d2, ND);                     // ds0
  gemm_wmma_kernel<1,0,0><<<gg(KD,KD), b256, 0, stream>>>(h0, d2, nullptr, nullptr, gwh0, KD, KD, N);
  colsum_kernel<<<dim3(2), b256, 0, stream>>>(d2, gbh0, N, KD);
  gemm_wmma_kernel<0,1,3><<<gg(N,KD), b256, 0, stream>>>(d2, w_h0, nullptr, pr, d1, N, KD, KD); // dh0
  dsilu_kernel<<<eb(ND), b256, 0, stream>>>(d1, s_in, d2, ND);                   // ds_in
  gemm_wmma_kernel<1,0,0><<<gg(KD,KD), b256, 0, stream>>>(kn, d2, nullptr, nullptr, gwin, KD, KD, N);
  colsum_kernel<<<dim3(2), b256, 0, stream>>>(d2, gbin, N, KD);

  // ---- stage 5: parameter update: p*(1-alpha) + eta*mom - theta*g
  update_kernel<<<eb(PW), b256, 0, stream>>>(mem_w_in,  mom_w_in,      gwin,  scal, nwin,  PW);
  update_kernel<<<eb(KD), b256, 0, stream>>>(mem_b_in,  mom_b_in,      gbin,  scal, nbin,  KD);
  update_kernel<<<eb(PW), b256, 0, stream>>>(w_h0,      mom_w_h,       gwh0,  scal, nwh0,  PW);
  update_kernel<<<eb(PW), b256, 0, stream>>>(w_h1,      mom_w_h + PW,  gwh1,  scal, nwh1,  PW);
  update_kernel<<<eb(KD), b256, 0, stream>>>(b_h0,      mom_b_h,       gbh0,  scal, nbh0,  KD);
  update_kernel<<<eb(KD), b256, 0, stream>>>(b_h1,      mom_b_h + KD,  gbh1,  scal, nbh1,  KD);
  update_kernel<<<eb(PW), b256, 0, stream>>>(mem_w_out, mom_w_out,     gwout, scal, nwout, PW);
  update_kernel<<<eb(KD), b256, 0, stream>>>(mem_b_out, mom_b_out,     gbout, scal, nbout, KD);

  // ---- stage 6: retrieval = silu(mlp(new_params, q))
  gemm_wmma_kernel<0,0,1><<<gg(N,KD), b256, 0, stream>>>(qn, nwin,  nbin,  nullptr, h0, N, KD, KD);
  gemm_wmma_kernel<0,0,2><<<gg(N,KD), b256, 0, stream>>>(h0, nwh0,  nbh0,  h0,      h1, N, KD, KD);
  gemm_wmma_kernel<0,0,2><<<gg(N,KD), b256, 0, stream>>>(h1, nwh1,  nbh1,  h1,      h2, N, KD, KD);
  gemm_wmma_kernel<0,0,1><<<gg(N,KD), b256, 0, stream>>>(h2, nwout, nbout, nullptr, pr, N, KD, KD);

  // ---- stage 7: sliding-window attention + output projection
  gemm_wmma_kernel<0,0,0><<<gg(N,KD), b256, 0, stream>>>(pr, swa_wq, swa_bq, nullptr, s_in, N, KD, KD);
  gemm_wmma_kernel<0,0,0><<<gg(N,KD), b256, 0, stream>>>(pr, swa_wk, swa_bk, nullptr, s0,   N, KD, KD);
  gemm_wmma_kernel<0,0,0><<<gg(N,KD), b256, 0, stream>>>(pr, swa_wv, swa_bv, nullptr, s1,   N, KD, KD);
  swa_attn_kernel<<<dim3(KT / 64, KNH, KB), b128, 0, stream>>>(s_in, s0, s1, d1, KT);
  gemm_wmma_kernel<0,0,0><<<gg(N,KD), b256, 0, stream>>>(d1, swa_wo, swa_bo, nullptr, d2, N, KD, KD);

  // ---- stage 8: drop meta tokens
  slice_out_kernel<<<eb((size_t)KB * KS * KD), b256, 0, stream>>>(d2, (float*)d_out);
}
